// MusicRegression_37280316129394
// MI455X (gfx1250) — matrix operand order repaired and checked by reading, not compiled, not run on hardware
//
#include <hip/hip_runtime.h>

typedef __attribute__((ext_vector_type(16))) _Float16 v16h;
typedef __attribute__((ext_vector_type(8)))  float    v8f;

#define SEQ 1024
#define DM  512
#define NH  8
#define DHD 64
#define DI  2048
#define NL  6
#define NB  4
#define OUTD 2

union FragU { uint4 q[2]; v16h v; };

__device__ __forceinline__ v8f wmma32(v16h a, v16h b, v8f c) {
  return __builtin_amdgcn_wmma_f32_16x16x32_f16(false, a, false, b, (short)0, c, false, false);
}

// ---------- fragment loaders (wave32; contiguous K per lane half, same in A and B) ----------
// A frag 16x32 (MxK) from row-major f16 (global or LDS). Per lane the 16 halves
// are contiguous: K = kh*16+0..15 (two 16B chunks).
__device__ __forceinline__ v16h load_a16(const _Float16* base, int lda) {
  int lane = threadIdx.x & 31;
  int m = lane & 15;
  int kh = lane >> 4;
  const _Float16* rp = base + (size_t)m * lda + kh * 16;
  FragU r;
  r.q[0] = *(const uint4*)rp;
  r.q[1] = *(const uint4*)(rp + 8);
  return r.v;
}

// B frag 32x16 (KxN): B[k][n] = src[(nbase+n)*ld + koff + k].
// Per lane: 16 contiguous halves (K = koff + kh*16 + 0..15) -> 2x b128.
__device__ __forceinline__ v16h load_b_k16(const _Float16* src, int ld, int nbase, int koff) {
  int lane = threadIdx.x & 31;
  int n = lane & 15;
  int kh = lane >> 4;
  const _Float16* rp = src + (size_t)(nbase + n) * ld + koff + kh * 16;
  FragU r;
  r.q[0] = *(const uint4*)rp;
  r.q[1] = *(const uint4*)(rp + 8);
  return r.v;
}

// B frag from f16 rel-pos table Eh[SEQ][DHD], row index clamped.
__device__ __forceinline__ v16h load_bE16(const _Float16* Eb, int rowbase, int koff) {
  int lane = threadIdx.x & 31;
  int n = lane & 15;
  int kh = lane >> 4;
  int row = rowbase + n;
  row = (row > SEQ - 1) ? (SEQ - 1) : row;
  const _Float16* rp = Eb + (size_t)row * DHD + koff + kh * 16;
  FragU r;
  r.q[0] = *(const uint4*)rp;
  r.q[1] = *(const uint4*)(rp + 8);
  return r.v;
}

// C/D frag store (f32): lane n = col; vgpr r = row rb+r (rb = 8*(lane>=16))
__device__ __forceinline__ void store_c_f32(float* C, int ldc, v8f c) {
  int lane = threadIdx.x & 31;
  int n = lane & 15;
  int rb = (lane >> 4) * 8;
#pragma unroll
  for (int r = 0; r < 8; ++r) C[(size_t)(rb + r) * ldc + n] = c[r];
}

// ---------------- kernels ----------------

__global__ __launch_bounds__(256) void embed_kernel(const int* __restrict__ x,
    const float* __restrict__ emb, const float* __restrict__ pos, float* __restrict__ h) {
  int idx = blockIdx.x * 256 + threadIdx.x;        // over NB*SEQ*DM
  int d = idx & (DM - 1);
  int bs = idx >> 9;
  int s = bs & (SEQ - 1);
  int tok = x[bs];
  h[idx] = emb[(size_t)tok * DM + d] * 22.62741699796952f + pos[(size_t)s * DM + d];
}

__global__ __launch_bounds__(256) void cvt_to_h16(const float* __restrict__ in,
    _Float16* __restrict__ out, int ntot) {
  int i = blockIdx.x * 256 + threadIdx.x;
  if (i < ntot) out[i] = (_Float16)in[i];
}

// transpose f32 [R][C] -> f16 [C][R]
__global__ __launch_bounds__(256) void transpose_f32_h16(const float* __restrict__ in,
    _Float16* __restrict__ out, int R, int C) {
  __shared__ float ts[32][33];
  int tx = threadIdx.x & 31, ty = threadIdx.x >> 5;
  int c0 = blockIdx.x * 32, r0 = blockIdx.y * 32;
#pragma unroll
  for (int i = 0; i < 32; i += 8)
    ts[ty + i][tx] = in[(size_t)(r0 + ty + i) * C + c0 + tx];
  __syncthreads();
#pragma unroll
  for (int i = 0; i < 32; i += 8)
    out[(size_t)(c0 + ty + i) * R + r0 + tx] = (_Float16)ts[tx][ty + i];
}

// batched transpose f16 [R][C] -> f16 [C][R] (blockIdx.z = batch)
__global__ __launch_bounds__(256) void transpose_h16(const _Float16* __restrict__ in,
    _Float16* __restrict__ out, int R, int C) {
  __shared__ _Float16 ts[32][33];
  const _Float16* ip = in + (size_t)blockIdx.z * R * C;
  _Float16* op = out + (size_t)blockIdx.z * R * C;
  int tx = threadIdx.x & 31, ty = threadIdx.x >> 5;
  int c0 = blockIdx.x * 32, r0 = blockIdx.y * 32;
#pragma unroll
  for (int i = 0; i < 32; i += 8)
    ts[ty + i][tx] = ip[(size_t)(r0 + ty + i) * C + c0 + tx];
  __syncthreads();
#pragma unroll
  for (int i = 0; i < 32; i += 8)
    op[(size_t)(c0 + ty + i) * R + r0 + tx] = ts[tx][ty + i];
}

// C[M,N] = act(A @ Wt^T + bias). A: f32 (Af) or f16 (Ah) selected by Ah!=null.
// Wt pre-transposed f16 [N][K]. Block tile 64x256, 8 waves of 32x64,
// K staged 64 at a time (16 WMMAs per barrier pair).
__global__ __launch_bounds__(256) void gemm_bias3(const float* __restrict__ Af,
    const _Float16* __restrict__ Ah, const _Float16* __restrict__ Wt,
    const float* __restrict__ bias, float* __restrict__ Cf, _Float16* __restrict__ Ch,
    int Ndim, int Kdim, int relu, int out16) {
  __shared__ _Float16 As[64][72];
  int tid = threadIdx.x, wv = tid >> 5;
  int bm = blockIdx.x * 64, bn = blockIdx.y * 256;
  int wm = (wv >> 2) * 32, wn = (wv & 3) * 64;
  v8f acc[2][4] = {};
  int arow = tid >> 2, akc = (tid & 3) * 16;
  const int nb0 = bn + wn;
  for (int k0 = 0; k0 < Kdim; k0 += 64) {
    // stage A block 64x64 f16
    if (Ah) {
      const uint4* ap = (const uint4*)(Ah + (size_t)(bm + arow) * Kdim + k0 + akc);
      uint4 u0 = ap[0], u1 = ap[1];
      *(uint4*)&As[arow][akc]     = u0;
      *(uint4*)&As[arow][akc + 8] = u1;
      if (k0 + 64 < Kdim)
        __builtin_prefetch((const void*)(Ah + (size_t)(bm + arow) * Kdim + k0 + 64 + akc), 0, 1);
    } else {
      const float4* ap = (const float4*)(Af + (size_t)(bm + arow) * Kdim + k0 + akc);
      float4 a0 = ap[0], a1 = ap[1], a2 = ap[2], a3 = ap[3];
      union { uint4 q[2]; _Float16 hh[16]; } pk;
      pk.hh[0]  = (_Float16)a0.x; pk.hh[1]  = (_Float16)a0.y;
      pk.hh[2]  = (_Float16)a0.z; pk.hh[3]  = (_Float16)a0.w;
      pk.hh[4]  = (_Float16)a1.x; pk.hh[5]  = (_Float16)a1.y;
      pk.hh[6]  = (_Float16)a1.z; pk.hh[7]  = (_Float16)a1.w;
      pk.hh[8]  = (_Float16)a2.x; pk.hh[9]  = (_Float16)a2.y;
      pk.hh[10] = (_Float16)a2.z; pk.hh[11] = (_Float16)a2.w;
      pk.hh[12] = (_Float16)a3.x; pk.hh[13] = (_Float16)a3.y;
      pk.hh[14] = (_Float16)a3.z; pk.hh[15] = (_Float16)a3.w;
      *(uint4*)&As[arow][akc]     = pk.q[0];
      *(uint4*)&As[arow][akc + 8] = pk.q[1];
      if (k0 + 64 < Kdim)
        __builtin_prefetch((const void*)(Af + (size_t)(bm + arow) * Kdim + k0 + 64 + akc), 0, 1);
    }
    __syncthreads();
#pragma unroll
    for (int kk = 0; kk < 64; kk += 32) {
      v16h fa0 = load_a16(&As[wm][kk], 72);
      v16h fa1 = load_a16(&As[wm + 16][kk], 72);
      v16h fb0 = load_b_k16(Wt, Kdim, nb0,      k0 + kk);
      v16h fb1 = load_b_k16(Wt, Kdim, nb0 + 16, k0 + kk);
      v16h fb2 = load_b_k16(Wt, Kdim, nb0 + 32, k0 + kk);
      v16h fb3 = load_b_k16(Wt, Kdim, nb0 + 48, k0 + kk);
      acc[0][0] = wmma32(fa0, fb0, acc[0][0]);
      acc[0][1] = wmma32(fa0, fb1, acc[0][1]);
      acc[0][2] = wmma32(fa0, fb2, acc[0][2]);
      acc[0][3] = wmma32(fa0, fb3, acc[0][3]);
      acc[1][0] = wmma32(fa1, fb0, acc[1][0]);
      acc[1][1] = wmma32(fa1, fb1, acc[1][1]);
      acc[1][2] = wmma32(fa1, fb2, acc[1][2]);
      acc[1][3] = wmma32(fa1, fb3, acc[1][3]);
    }
    __syncthreads();
  }
  int lane = tid & 31, n = lane & 15, rb = (lane >> 4) * 8;
#pragma unroll
  for (int ni = 0; ni < 4; ++ni) {
    int col = nb0 + ni * 16 + n;
    float bi = bias[col];
#pragma unroll
    for (int mi = 0; mi < 2; ++mi) {
      int row = bm + wm + mi * 16 + rb;
#pragma unroll
      for (int r = 0; r < 8; ++r) {
        float v = acc[mi][ni][r] + bi;
        if (relu) v = fmaxf(v, 0.f);
        if (out16) Ch[(size_t)(row + r) * Ndim + col] = (_Float16)v;
        else       Cf[(size_t)(row + r) * Ndim + col] = v;
      }
    }
  }
}

// Fused rel-pos flash attention, all-f16 operands, f16 output.
// grid: (NB*NH, SEQ/128), block 256 (8 waves x 16 q-rows).
__global__ __launch_bounds__(256) void flash_attn(const _Float16* __restrict__ Q,
    const _Float16* __restrict__ Km, const _Float16* __restrict__ Vt,
    const _Float16* __restrict__ Eh, _Float16* __restrict__ O16) {
  __shared__ float    Tl[8][16][34];
  __shared__ _Float16 Pl[8][16][48];
  int tid = threadIdx.x, lane = tid & 31, wv = tid >> 5;
  int b = blockIdx.x >> 3, hh = blockIdx.x & 7;
  int i0 = blockIdx.y * 128 + wv * 16;
  int n = lane & 15, rb = (lane >> 4) * 8;

  const _Float16* qp = Q + ((size_t)(b * SEQ + i0) * DM + hh * DHD);
  v16h qa0 = load_a16(qp, DM);
  v16h qa1 = load_a16(qp + 32, DM);
  const _Float16* kbase = Km + ((size_t)b * SEQ) * DM + hh * DHD;
  const _Float16* vtb = Vt + ((size_t)b * DM + hh * DHD) * SEQ;

  float m8[8], l8[8];
#pragma unroll
  for (int r = 0; r < 8; ++r) { m8[r] = -3.0e38f; l8[r] = 0.f; }
  v8f o0 = {}, o1 = {}, o2 = {}, o3 = {};

  // zero the K=16..31 pad of this wave's P tile (per-wave region, same-wave use)
  {
    _Float16* p = &Pl[wv][0][0];
    int row = lane >> 1, c0 = 16 + (lane & 1) * 8;
#pragma unroll
    for (int z = 0; z < 8; ++z) p[row * 48 + c0 + z] = (_Float16)0.f;
  }

  auto tile = [&](int j0, bool do_srel) {
    const _Float16* kp = kbase + (size_t)j0 * DM;
    v8f s = {};
    s = wmma32(qa0, load_b_k16(kp, DM, 0, 0), s);
    s = wmma32(qa1, load_b_k16(kp, DM, 0, 32), s);
    if (do_srel) {
      // Srel[i,j] = q_i . E[SEQ-1-i+j], nonzero only for j<=i
      int m0 = SEQ - 16 - i0 + j0;   // >= 0 since i0 <= SEQ-16
      v8f t0 = {}, t1 = {};
      t0 = wmma32(qa0, load_bE16(Eh, m0, 0), t0);
      t0 = wmma32(qa1, load_bE16(Eh, m0, 32), t0);
      t1 = wmma32(qa0, load_bE16(Eh, m0 + 16, 0), t1);
      t1 = wmma32(qa1, load_bE16(Eh, m0 + 16, 32), t1);
      float* tb = &Tl[wv][0][0];
      store_c_f32(tb, 34, t0);
      store_c_f32(tb + 16, 34, t1);
#pragma unroll
      for (int r = 0; r < 8; ++r) {
        int ii = rb + r;
        if (j0 + n <= i0 + ii) s[r] += tb[ii * 34 + (15 - ii + n)];
      }
    }
    _Float16* pb = &Pl[wv][0][0];
#pragma unroll
    for (int r = 0; r < 8; ++r) {
      float lg = s[r] * 0.125f;   // 1/sqrt(64)
      float mx = lg;
      mx = fmaxf(mx, __shfl_xor(mx, 1, 32));
      mx = fmaxf(mx, __shfl_xor(mx, 2, 32));
      mx = fmaxf(mx, __shfl_xor(mx, 4, 32));
      mx = fmaxf(mx, __shfl_xor(mx, 8, 32));
      float mn = fmaxf(m8[r], mx);
      float al = __expf(m8[r] - mn);
      float p  = __expf(lg - mn);
      float rs = p;
      rs += __shfl_xor(rs, 1, 32);
      rs += __shfl_xor(rs, 2, 32);
      rs += __shfl_xor(rs, 4, 32);
      rs += __shfl_xor(rs, 8, 32);
      l8[r] = l8[r] * al + rs;
      m8[r] = mn;
      o0[r] *= al; o1[r] *= al; o2[r] *= al; o3[r] *= al;
      pb[(rb + r) * 48 + n] = (_Float16)p;
    }
    v16h pa = load_a16(pb, 48);
    o0 = wmma32(pa, load_b_k16(vtb, SEQ,  0, j0), o0);
    o1 = wmma32(pa, load_b_k16(vtb, SEQ, 16, j0), o1);
    o2 = wmma32(pa, load_b_k16(vtb, SEQ, 32, j0), o2);
    o3 = wmma32(pa, load_b_k16(vtb, SEQ, 48, j0), o3);
  };

  for (int j0 = 0; j0 <= i0; j0 += 16)       tile(j0, true);   // Srel region
  for (int j0 = i0 + 16; j0 < SEQ; j0 += 16) tile(j0, false);  // pure QK region

  _Float16* op = O16 + ((size_t)(b * SEQ + i0 + rb) * DM + hh * DHD) + n;
#pragma unroll
  for (int r = 0; r < 8; ++r) {
    float inv = 1.0f / l8[r];
    op[(size_t)r * DM +  0] = (_Float16)(o0[r] * inv);
    op[(size_t)r * DM + 16] = (_Float16)(o1[r] * inv);
    op[(size_t)r * DM + 32] = (_Float16)(o2[r] * inv);
    op[(size_t)r * DM + 48] = (_Float16)(o3[r] * inv);
  }
}

// out[row] = LN(a[row] + res[row]) * sc + bi ; one block per row of 512
__global__ __launch_bounds__(256) void add_ln_kernel(const float* __restrict__ a,
    const float* __restrict__ res, const float* __restrict__ sc,
    const float* __restrict__ bi, float* __restrict__ out) {
  __shared__ float red[256];
  int row = blockIdx.x, t = threadIdx.x;
  const float* ap = a + (size_t)row * DM;
  const float* rp = res + (size_t)row * DM;
  float x0 = ap[t] + rp[t];
  float x1 = ap[t + 256] + rp[t + 256];
  red[t] = x0 + x1;
  __syncthreads();
  for (int s2 = 128; s2 > 0; s2 >>= 1) {
    if (t < s2) red[t] += red[t + s2];
    __syncthreads();
  }
  float mu = red[0] * (1.0f / DM);
  __syncthreads();
  float d0 = x0 - mu, d1 = x1 - mu;
  red[t] = d0 * d0 + d1 * d1;
  __syncthreads();
  for (int s2 = 128; s2 > 0; s2 >>= 1) {
    if (t < s2) red[t] += red[t + s2];
    __syncthreads();
  }
  float inv = rsqrtf(red[0] * (1.0f / DM) + 1e-6f);
  out[(size_t)row * DM + t]       = d0 * inv * sc[t] + bi[t];
  out[(size_t)row * DM + t + 256] = d1 * inv * sc[t + 256] + bi[t + 256];
}

__global__ void final_kernel(const float* __restrict__ h, const float* __restrict__ Wf,
                             const float* __restrict__ bf, float* __restrict__ out) {
  int t = threadIdx.x;
  if (t < NB * OUTD) {
    int b = t >> 1, o = t & 1;
    float acc = bf[o];
    const float* hp = h + (size_t)b * SEQ * DM;   // row s=0
    for (int d = 0; d < DM; ++d) acc += hp[d] * Wf[d * OUTD + o];
    out[b * OUTD + o] = tanhf(acc);
  }
}

extern "C" void kernel_launch(void* const* d_in, const int* in_sizes, int n_in,
                              void* d_out, int out_size, void* d_ws, size_t ws_size,
                              hipStream_t stream) {
  const int*   x    = (const int*)d_in[0];
  const float* emb  = (const float*)d_in[1];
  const float* pos  = (const float*)d_in[2];
  const float* Wq   = (const float*)d_in[3];
  const float* bq   = (const float*)d_in[4];
  const float* Wk   = (const float*)d_in[5];
  const float* bk   = (const float*)d_in[6];
  const float* Wv   = (const float*)d_in[7];
  const float* bv   = (const float*)d_in[8];
  const float* Wo   = (const float*)d_in[9];
  const float* bo   = (const float*)d_in[10];
  const float* E    = (const float*)d_in[11];
  const float* W1   = (const float*)d_in[12];
  const float* b1   = (const float*)d_in[13];
  const float* W2   = (const float*)d_in[14];
  const float* b2   = (const float*)d_in[15];
  const float* l1s  = (const float*)d_in[16];
  const float* l1b  = (const float*)d_in[17];
  const float* l2s  = (const float*)d_in[18];
  const float* l2b  = (const float*)d_in[19];
  const float* Wf   = (const float*)d_in[20];
  const float* bf   = (const float*)d_in[21];

  const size_t BSD = (size_t)NB * SEQ * DM;            // 2,097,152
  const size_t WTL = 4 * (size_t)DM * DM + 2 * (size_t)DM * DI;  // halves per layer

  char* p = (char*)d_ws;
  auto alloc = [&](size_t bytes) { void* r = (void*)p; p += (bytes + 255) & ~(size_t)255; return r; };
  float*    h    = (float*)alloc(BSD * 4);
  float*    h1   = (float*)alloc(BSD * 4);
  float*    tmp  = (float*)alloc(BSD * 4);
  _Float16* q16  = (_Float16*)alloc(BSD * 2);
  _Float16* k16  = (_Float16*)alloc(BSD * 2);
  _Float16* v16b = (_Float16*)alloc(BSD * 2);
  _Float16* vt16 = (_Float16*)alloc(BSD * 2 + 128);    // tail pad for b128 overread
  _Float16* at16 = (_Float16*)alloc(BSD * 2);
  _Float16* mi16 = (_Float16*)alloc((size_t)NB * SEQ * DI * 2);
  _Float16* wts  = (_Float16*)alloc(WTL * NL * 2);
  _Float16* eh   = (_Float16*)alloc((size_t)NL * SEQ * DHD * 2);

  const int M = NB * SEQ;                              // 4096
  dim3 blk(256);
  dim3 gN512(M / 64, DM / 256);                        // (64, 2)
  dim3 gN2048(M / 64, DI / 256);                       // (64, 8)

  embed_kernel<<<(int)(BSD / 256), blk, 0, stream>>>(x, emb, pos, h);
  cvt_to_h16<<<(NL * SEQ * DHD) / 256, blk, 0, stream>>>(E, eh, NL * SEQ * DHD);

  // pre-transpose all weights to f16 [N][K]
  for (int l = 0; l < NL; ++l) {
    _Float16* wb = wts + (size_t)l * WTL;
    transpose_f32_h16<<<dim3(16, 16), blk, 0, stream>>>(Wq + (size_t)l * DM * DM, wb,           DM, DM);
    transpose_f32_h16<<<dim3(16, 16), blk, 0, stream>>>(Wk + (size_t)l * DM * DM, wb + 262144,  DM, DM);
    transpose_f32_h16<<<dim3(16, 16), blk, 0, stream>>>(Wv + (size_t)l * DM * DM, wb + 524288,  DM, DM);
    transpose_f32_h16<<<dim3(16, 16), blk, 0, stream>>>(Wo + (size_t)l * DM * DM, wb + 786432,  DM, DM);
    transpose_f32_h16<<<dim3(64, 16), blk, 0, stream>>>(W1 + (size_t)l * DM * DI, wb + 1048576, DM, DI);
    transpose_f32_h16<<<dim3(16, 64), blk, 0, stream>>>(W2 + (size_t)l * DI * DM, wb + 2097152, DI, DM);
  }

  for (int l = 0; l < NL; ++l) {
    _Float16* wqt = wts + (size_t)l * WTL;
    _Float16* wkt = wqt + 262144;
    _Float16* wvt = wqt + 524288;
    _Float16* wot = wqt + 786432;
    _Float16* w1t = wqt + 1048576;
    _Float16* w2t = wqt + 2097152;
    const _Float16* eh_l = eh + (size_t)l * SEQ * DHD;

    gemm_bias3<<<gN512, blk, 0, stream>>>(h, nullptr, wqt, bq + l * DM, nullptr, q16, DM, DM, 0, 1);
    gemm_bias3<<<gN512, blk, 0, stream>>>(h, nullptr, wkt, bk + l * DM, nullptr, k16, DM, DM, 0, 1);
    gemm_bias3<<<gN512, blk, 0, stream>>>(h, nullptr, wvt, bv + l * DM, nullptr, v16b, DM, DM, 0, 1);
    transpose_h16<<<dim3(DM / 32, SEQ / 32, NB), blk, 0, stream>>>(v16b, vt16, SEQ, DM);

    flash_attn<<<dim3(NB * NH, SEQ / 128), blk, 0, stream>>>(q16, k16, vt16, eh_l, at16);

    gemm_bias3<<<gN512, blk, 0, stream>>>(nullptr, at16, wot, bo + l * DM, tmp, nullptr, DM, DM, 0, 0);
    add_ln_kernel<<<M, blk, 0, stream>>>(tmp, h, l1s + l * DM, l1b + l * DM, h1);

    gemm_bias3<<<gN2048, blk, 0, stream>>>(h1, nullptr, w1t, b1 + l * DI, nullptr, mi16, DI, DM, 1, 1);
    gemm_bias3<<<gN512, blk, 0, stream>>>(nullptr, mi16, w2t, b2 + l * DM, tmp, nullptr, DM, DI, 0, 0);
    add_ln_kernel<<<M, blk, 0, stream>>>(tmp, h1, l2s + l * DM, l2b + l * DM, h);
  }

  final_kernel<<<1, 32, 0, stream>>>(h, Wf, bf, (float*)d_out);
}
